// _WindowAttention_893353197797
// MI455X (gfx1250) — matrix operand order repaired and checked by reading, not compiled, not run on hardware
//
#include <hip/hip_runtime.h>
#include <hip/hip_bf16.h>
#include <math.h>

// ---------------------------------------------------------------------------
// Fused Swin window attention for gfx1250 (MI455X), wave32 + v_wmma_f32_16x16x32_f16.
// One block per window-batch (2048 blocks x 256 threads = 8 waves).
// All intermediates in LDS (221 KB of 320 KB/WGP); weights stream from the
// 192 MB L2 with nt-major reuse; V is stored transposed so every WMMA fragment
// load is an aligned ds_load_b128 pair.
// ---------------------------------------------------------------------------

typedef __attribute__((ext_vector_type(16))) _Float16 v16h;
typedef __attribute__((ext_vector_type(8)))  _Float16 v8h;
typedef __attribute__((ext_vector_type(8)))  float    v8f;

#define NUM_HEADS 6
#define DIM   192
#define NWIN  256
#define NTOK  65
#define NPAD  80          // 65 tokens padded to 5 row-tiles of 16
#define QKD   384         // q|k columns kept in LDS
#define HD    32
#define SROW  96          // attention row / Vt row padded to 3 K-chunks of 32
#define QSCALE 0.17677669529663687f   // 1/sqrt(32)

// LDS layout (bytes); every base is 16B-aligned
#define OFF_QK   0                                     // 80x384 f16 = 61440 B (q|k)
#define OFF_S    (NPAD * QKD * 2)                      // 6x80x96 f16 = 92160 B
#define OFF_X    (OFF_S + NUM_HEADS * NPAD * SROW * 2) // 80x192 f16 = 30720 B (x tile / mask / attn-out)
#define OFF_V    (OFF_X + NPAD * DIM * 2)              // 192x96 f16 = 36864 B (V^T)
#define SMEM_BYTES (OFF_V + DIM * SROW * 2)            // 221184 B total

// ---- A fragment (16x32 f16, sequential K per lane-row) from LDS row-major f16
__device__ __forceinline__ v16h fragA_lds(const _Float16* base, int row0, int stride,
                                          int k0, int lane) {
  int m = lane & 15, half = lane >> 4;
  const _Float16* p = base + (row0 + m) * stride + k0 + 16 * half;
  v8h lo = *(const v8h*)p;          // K = k0 + 16*half + 0..7
  v8h hi = *(const v8h*)(p + 8);    // K = k0 + 16*half + 8..15
  v16h r;
#pragma unroll
  for (int i = 0; i < 8; ++i) { r[i] = lo[i]; r[i + 8] = hi[i]; }
  return r;
}

// ---- B fragment (32x16 f16, sequential K per lane-column): column n = source row n
__device__ __forceinline__ v16h fragB_lds(const _Float16* base, int row0, int stride,
                                          int k0, int lane) {
  int n = lane & 15, half = lane >> 4;
  const _Float16* p = base + (row0 + n) * stride + k0 + 16 * half;
  v8h lo = *(const v8h*)p;          // K = k0 + 16*half + 0..7
  v8h hi = *(const v8h*)(p + 8);    // K = k0 + 16*half + 8..15
  v16h r;
#pragma unroll
  for (int i = 0; i < 8; ++i) { r[i] = lo[i]; r[i + 8] = hi[i]; }
  return r;
}

// ---- B fragment from global f32 weight rows (out = act @ W^T => B col n = W row n),
//      converted f32->f16 in flight; weights are L2-resident (192 MB L2).
__device__ __forceinline__ v16h fragB_gf32(const float* __restrict__ w, int row0,
                                           int stride, int k0, int lane) {
  int n = lane & 15, half = lane >> 4;
  const float* p = w + (size_t)(row0 + n) * stride + k0 + 16 * half;
  v16h r;
#pragma unroll
  for (int e = 0; e < 16; ++e) r[e] = (_Float16)p[e];
  return r;
}

__device__ __forceinline__ v8f wmma_f16(v16h a, v16h b, v8f c) {
  return __builtin_amdgcn_wmma_f32_16x16x32_f16(false, a, false, b, (short)0, c,
                                                false, false);
}

__global__ void __launch_bounds__(256)
win_attn_fused(const float* __restrict__ x, const float* __restrict__ mask,
               const float* __restrict__ qkv_w, const float* __restrict__ qkv_b,
               const float* __restrict__ bias_table,
               const float* __restrict__ proj_w, const float* __restrict__ proj_b,
               float* __restrict__ out) {
  extern __shared__ __align__(16) char smem[];
  _Float16* qk = (_Float16*)(smem + OFF_QK);   // 80 x 384 (q pre-scaled | k)
  _Float16* Sh = (_Float16*)(smem + OFF_S);    // 6 x 80 x 96 attention logits/probs
  _Float16* xh = (_Float16*)(smem + OFF_X);    // 80 x 192: x tile -> mask -> attn-out
  _Float16* Vt = (_Float16*)(smem + OFF_V);    // 192 x 96: V transposed (k-major)

  const int tid  = threadIdx.x;
  const int lane = tid & 31;
  const int wave = tid >> 5;
  const int b    = blockIdx.x;            // window-batch 0..2047
  const int wwin = b & (NWIN - 1);        // mask window index (b = outer*nW + w)
  const int nn   = lane & 15;
  const int half = lane >> 4;

  // ---------------- Phase 0: stage x tile (f32->f16, zero-pad rows 65..79) and
  //                  zero-fill Vt's K padding (k = 80..95)
  {
    const float* xg = x + (size_t)b * NTOK * DIM;
    for (int idx = tid; idx < NPAD * DIM; idx += 256) {
      int r = idx / DIM, c = idx - r * DIM;
      xh[idx] = (r < NTOK) ? (_Float16)xg[r * DIM + c] : (_Float16)0.0f;
    }
    for (int idx = tid; idx < DIM * 16; idx += 256) {
      int vc = idx >> 4, k = NPAD + (idx & 15);
      Vt[vc * SROW + k] = (_Float16)0.0f;
    }
  }
  __syncthreads();

  // ---------------- Phase 1: QKV GEMM, nt-major so weight fragments are loaded
  //                  and converted once per column tile and reused over 5 row tiles.
  for (int nt = wave; nt < 36; nt += 8) {           // wave-uniform: EXEC stays full
    v16h bw[6];
#pragma unroll
    for (int kc = 0; kc < 6; ++kc)
      bw[kc] = fragB_gf32(qkv_w, nt * 16, DIM, kc * 32, lane);
    const int ng = nt * 16 + nn;
    const float bn = qkv_b[ng];
    const bool isV = (ng >= 2 * DIM);
#pragma unroll
    for (int mt = 0; mt < 5; ++mt) {
      v8f acc = {0.f, 0.f, 0.f, 0.f, 0.f, 0.f, 0.f, 0.f};
#pragma unroll
      for (int kc = 0; kc < 6; ++kc)
        acc = wmma_f16(fragA_lds(xh, mt * 16, DIM, kc * 32, lane), bw[kc], acc);
#pragma unroll
      for (int r = 0; r < 8; ++r) {
        int m = mt * 16 + r + 8 * half;
        float v = acc[r] + bn;
        if (ng < DIM) v *= QSCALE;                  // scale q only
        if (isV) Vt[(ng - 2 * DIM) * SROW + m] = (_Float16)v;   // V stored transposed
        else     qk[m * QKD + ng] = (_Float16)v;
      }
    }
  }
  __syncthreads();

  // ---------------- Phase 2a: S = q @ k^T per head; A (q rows) reused across nt.
  //                  Concurrently stage this window's 64x64 mask into the idle x region.
  for (int t = wave; t < NUM_HEADS * 5; t += 8) {
    int h = t / 5, mt = t - h * 5;
    v16h a = fragA_lds(qk, mt * 16, QKD, h * HD, lane);
#pragma unroll
    for (int nt = 0; nt < 5; ++nt) {
      v16h bb = fragB_lds(qk, nt * 16, QKD, DIM + h * HD, lane);
      v8f acc = {0.f, 0.f, 0.f, 0.f, 0.f, 0.f, 0.f, 0.f};
      acc = wmma_f16(a, bb, acc);
#pragma unroll
      for (int r = 0; r < 8; ++r) {
        int m = mt * 16 + r + 8 * half;
        Sh[(h * NPAD + m) * SROW + nt * 16 + nn] = (_Float16)acc[r];
      }
    }
  }
  {
    float* maskL = (float*)xh;                      // x tile is dead; reuse for mask
    const float* mg = mask + (size_t)wwin * 4096;
    for (int idx = tid; idx < 4096; idx += 256) maskL[idx] = mg[idx];
  }
  __syncthreads();

  // ---------------- Phase 2b: + rel-pos bias + window mask (from LDS), row softmax
  {
    const float* maskL = (const float*)xh;
    for (int rid = tid; rid < NUM_HEADS * NTOK; rid += 256) {
      int h = rid / NTOK, i = rid - h * NTOK;
      _Float16* row = Sh + (h * NPAD + i) * SROW;
      float mx = -3.0e38f;
      for (int j = 0; j < NTOK; ++j) {
        float v = (float)row[j];
        if (i > 0 && j > 0) {               // token row/col gets zero-padded bias/mask
          int p = i - 1, q = j - 1;
          int d0 = (p >> 3) - (q >> 3) + 7; // rel index from static 8x8 geometry
          int d1 = (p & 7) - (q & 7) + 7;
          v += bias_table[(d0 * 15 + d1) * NUM_HEADS + h] + maskL[p * 64 + q];
        }
        row[j] = (_Float16)v;
        mx = fmaxf(mx, v);
      }
      float s = 0.f;
      for (int j = 0; j < NTOK; ++j) {
        float e = __expf((float)row[j] - mx);
        s += e;
        row[j] = (_Float16)e;
      }
      float inv = 1.0f / s;
      for (int j = 0; j < NTOK; ++j) row[j] = (_Float16)((float)row[j] * inv);
      for (int j = NTOK; j < SROW; ++j) row[j] = (_Float16)0.0f;  // kill K padding
    }
  }
  __syncthreads();

  // ---------------- Phase 3: attn-out = P @ V^T-rows; A (P rows) reused across nt
  for (int t = wave; t < NUM_HEADS * 5; t += 8) {
    int h = t / 5, mt = t - h * 5;
    v16h a0 = fragA_lds(Sh + h * NPAD * SROW, mt * 16, SROW, 0, lane);
    v16h a1 = fragA_lds(Sh + h * NPAD * SROW, mt * 16, SROW, 32, lane);
    v16h a2 = fragA_lds(Sh + h * NPAD * SROW, mt * 16, SROW, 64, lane);
#pragma unroll
    for (int nt = 0; nt < 2; ++nt) {
      v8f acc = {0.f, 0.f, 0.f, 0.f, 0.f, 0.f, 0.f, 0.f};
      acc = wmma_f16(a0, fragB_lds(Vt, h * HD + nt * 16, SROW, 0, lane), acc);
      acc = wmma_f16(a1, fragB_lds(Vt, h * HD + nt * 16, SROW, 32, lane), acc);
      acc = wmma_f16(a2, fragB_lds(Vt, h * HD + nt * 16, SROW, 64, lane), acc);
      int col = h * HD + nt * 16 + nn;
#pragma unroll
      for (int r = 0; r < 8; ++r) {
        int m = mt * 16 + r + 8 * half;
        xh[m * DIM + col] = (_Float16)acc[r];      // xh now holds attn-out tile
      }
    }
  }
  __syncthreads();

  // ---------------- Phase 4: projection (80x192) @ (192x192)^T-rows + bias, store
  for (int t = wave; t < 5 * 12; t += 8) {
    int mt = t / 12, nt = t - mt * 12;
    v8f acc = {0.f, 0.f, 0.f, 0.f, 0.f, 0.f, 0.f, 0.f};
#pragma unroll
    for (int kc = 0; kc < 6; ++kc) {
      v16h a  = fragA_lds(xh, mt * 16, DIM, kc * 32, lane);
      v16h bb = fragB_gf32(proj_w, nt * 16, DIM, kc * 32, lane);
      acc = wmma_f16(a, bb, acc);
    }
    int ng = nt * 16 + nn;
    float bp = proj_b[ng];
    float* og = out + (size_t)b * NTOK * DIM;
#pragma unroll
    for (int r = 0; r < 8; ++r) {
      int m = mt * 16 + r + 8 * half;
      if (m < NTOK) og[m * DIM + ng] = acc[r] + bp;   // drop pad rows
    }
  }
}

extern "C" void kernel_launch(void* const* d_in, const int* in_sizes, int n_in,
                              void* d_out, int out_size, void* d_ws, size_t ws_size,
                              hipStream_t stream) {
  (void)in_sizes; (void)n_in; (void)d_ws; (void)ws_size; (void)out_size;
  const float* x          = (const float*)d_in[0];
  const float* mask       = (const float*)d_in[1];
  const float* qkv_w      = (const float*)d_in[2];
  const float* qkv_b      = (const float*)d_in[3];
  const float* bias_table = (const float*)d_in[4];
  const float* proj_w     = (const float*)d_in[5];
  const float* proj_b     = (const float*)d_in[6];
  // d_in[7]=add_token, d_in[8]=token_num are static (1,1) in this config.
  float* out = (float*)d_out;

  dim3 grid(2048), block(256);
  win_attn_fused<<<grid, block, SMEM_BYTES, stream>>>(
      x, mask, qkv_w, qkv_b, bias_table, proj_w, proj_b, out);
}
